// StackedAttention_18528488915475
// MI455X (gfx1250) — compile-verified, matrix operand order checked by reading, not run on hardware
//
#include <hip/hip_runtime.h>
#include <stdint.h>

typedef __attribute__((ext_vector_type(16))) _Float16 v16h;
typedef __attribute__((ext_vector_type(8)))  _Float16 v8h;
typedef __attribute__((ext_vector_type(8)))  float    v8f;

#define B_DIM 4
#define S_DIM 4096
#define E_DIM 64
#define XROW 4097   // S+1

// ---------------------------------------------------------------------------
// WMMA helpers (CDNA5 wave32, V_WMMA_F32_16X16X32_F16)
// A frag (16x32 f16): lane = row(0..15)+16*sel; elems 0..7 -> K=k0+sel*8+i,
//                     elems 8..15 -> K=k0+16+sel*8+i
// B frag (32x16 f16): lane = col(0..15)+16*sel; same K mapping as A
// C/D frag (16x16 f32): VGPR r, lane: row = r + 8*sel, col = lane&15
// ---------------------------------------------------------------------------
__device__ __forceinline__ v8f wmma_f16(v16h a, v16h b, v8f c) {
  return __builtin_amdgcn_wmma_f32_16x16x32_f16(false, a, false, b, (short)0, c,
                                                false, false);
}

__device__ __forceinline__ v16h load_frag16(const _Float16* rowbase, int k0, int sel) {
  const v8h lo = *(const v8h*)(rowbase + k0 + sel * 8);
  const v8h hi = *(const v8h*)(rowbase + k0 + 16 + sel * 8);
  v16h r;
#pragma unroll
  for (int i = 0; i < 8; ++i) { r[i] = lo[i]; r[8 + i] = hi[i]; }
  return r;
}

// 16-byte async global->LDS copy, tracked by ASYNCcnt
__device__ __forceinline__ void async_copy16(uint32_t lds_off, uint64_t gaddr) {
  asm volatile("global_load_async_to_lds_b128 %0, %1, off"
               :: "v"(lds_off), "v"(gaddr) : "memory");
}
__device__ __forceinline__ void wait_async0() {
  asm volatile("s_wait_asynccnt 0x0" ::: "memory");
}
__device__ __forceinline__ uint32_t lds_offset(const void* p) {
  return (uint32_t)(uintptr_t)p;  // generic LDS ptr: addr[31:0] == LDS offset
}

// ---------------------------------------------------------------------------
// Kernel 0: prep — rmT[64][4096] (f16, transposed, col63 zero-padded) and
//                  wT[l][m][d][e] (f16, transposed weights)
// ---------------------------------------------------------------------------
__global__ void prep_kernel(const float* __restrict__ rm,
                            const float* __restrict__ qw,
                            const float* __restrict__ kw,
                            const float* __restrict__ vw,
                            _Float16* __restrict__ rmT,
                            _Float16* __restrict__ wT) {
  int i = blockIdx.x * 256 + threadIdx.x;
  if (i < 64 * 4096) {
    int e = i >> 12;
    int k = i & 4095;
    rmT[i] = (e < 63) ? (_Float16)rm[k * 63 + e] : (_Float16)0.0f;
  }
  if (i < 3 * 3 * 64 * 64) {  // [l][m][d][e]
    int l = i / (3 * 4096);
    int rem = i % (3 * 4096);
    int m = rem >> 12;
    int d = (rem >> 6) & 63;
    int e = rem & 63;
    const float* src = (m == 0) ? qw : (m == 1) ? kw : vw;
    wT[i] = (_Float16)src[l * 4096 + e * 64 + d];
  }
}

// ---------------------------------------------------------------------------
// Kernel 1: projection, K split 4 ways across waves for occupancy.
// grid 512 x 256; deterministic LDS tree reduction; col 63 = x pass-through.
// ---------------------------------------------------------------------------
__global__ void proj_kernel(const float* __restrict__ x,
                            const _Float16* __restrict__ rmT,
                            float* __restrict__ av) {
  __shared__ float partial[8][1024];      // 32 KB
  int wid = threadIdx.x >> 5;
  int lane = threadIdx.x & 31;
  int ln = lane & 15;
  int sel = lane >> 4;
  int job = blockIdx.x * 8 + wid;         // 0..4095
  int tile = job >> 2;                    // 0..1023 (16-row tiles)
  int kc = job & 3;                       // K chunk
  int b = tile >> 8;
  int s0 = (tile & 255) * 16;

  const float* xrow = x + (size_t)(b * S_DIM + s0 + ln) * XROW;
  v8f acc[4] = {};
  int kbeg = kc * 1024;
  int kend = kbeg + 1024;

  for (int k0 = kbeg; k0 < kend; k0 += 32) {
    __builtin_prefetch(xrow + k0 + 256, 0, 0);  // speculative stream-ahead
    v16h a;
    int kA = k0 + sel * 8;
    int kB = k0 + 16 + sel * 8;
#pragma unroll
    for (int i = 0; i < 8; ++i) {
      a[i]     = (_Float16)xrow[kA + i];
      a[8 + i] = (_Float16)xrow[kB + i];
    }
#pragma unroll
    for (int nt = 0; nt < 4; ++nt) {
      const _Float16* brow = rmT + (size_t)(nt * 16 + ln) * S_DIM;
      v16h bf = load_frag16(brow, k0, sel);
      acc[nt] = wmma_f16(a, bf, acc[nt]);
    }
  }

  float* pw = partial[wid];
#pragma unroll
  for (int nt = 0; nt < 4; ++nt) {
    int col = nt * 16 + ln;
#pragma unroll
    for (int r = 0; r < 8; ++r) pw[(r + 8 * sel) * 64 + col] = acc[nt][r];
  }
  __syncthreads();

  int tid = threadIdx.x;
#pragma unroll
  for (int t = 0; t < 2; ++t) {
    int tl = blockIdx.x * 2 + t;
    int tb = tl >> 8;
    int ts0 = (tl & 255) * 16;
    float* avbase = av + (size_t)(tb * S_DIM + ts0) * E_DIM;
    for (int i = tid; i < 1024; i += 256) {
      float v = (partial[t * 4 + 0][i] + partial[t * 4 + 1][i]) +
                (partial[t * 4 + 2][i] + partial[t * 4 + 3][i]);
      int col = i & 63;
      int row = i >> 6;
      if (col == 63)
        v = x[(size_t)(tb * S_DIM + ts0 + row) * XROW + (XROW - 1)];
      avbase[i] = v;
    }
  }
}

// ---------------------------------------------------------------------------
// Kernel 2: QKV GEMMs for one layer. Q,K row-major f16 [b][s][d]; Vt[b][d][s].
// ---------------------------------------------------------------------------
__global__ void qkv_kernel(const float* __restrict__ av,
                           const _Float16* __restrict__ wT, int layer,
                           _Float16* __restrict__ Qb,
                           _Float16* __restrict__ Kb,
                           _Float16* __restrict__ Vtb) {
  int wid = threadIdx.x >> 5;
  int lane = threadIdx.x & 31;
  int ln = lane & 15;
  int sel = lane >> 4;
  int gtile = blockIdx.x * 8 + wid;
  int b = gtile >> 8;
  int s0 = (gtile & 255) * 16;

  const float* avrow = av + (size_t)(b * S_DIM + s0 + ln) * E_DIM;
  v16h a0, a1;
#pragma unroll
  for (int i = 0; i < 8; ++i) {
    a0[i]     = (_Float16)avrow[sel * 8 + i];
    a0[8 + i] = (_Float16)avrow[16 + sel * 8 + i];
    a1[i]     = (_Float16)avrow[32 + sel * 8 + i];
    a1[8 + i] = (_Float16)avrow[48 + sel * 8 + i];
  }

  for (int m = 0; m < 3; ++m) {
    const _Float16* w = wT + (size_t)(layer * 3 + m) * 4096;  // [d][e]
    v8f acc[4] = {};
#pragma unroll
    for (int nt = 0; nt < 4; ++nt) {
      const _Float16* wrow = w + (size_t)(nt * 16 + ln) * E_DIM;
      v16h b0 = load_frag16(wrow, 0, sel);
      v16h b1 = load_frag16(wrow, 32, sel);
      acc[nt] = wmma_f16(a0, b0, acc[nt]);
      acc[nt] = wmma_f16(a1, b1, acc[nt]);
    }
#pragma unroll
    for (int nt = 0; nt < 4; ++nt) {
      int col = nt * 16 + ln;
#pragma unroll
      for (int r = 0; r < 8; ++r) {
        int rr = r + 8 * sel;
        _Float16 h = (_Float16)acc[nt][r];
        size_t srow = (size_t)(b * S_DIM + s0 + rr);
        if (m == 0)      Qb[srow * E_DIM + col] = h;
        else if (m == 1) Kb[srow * E_DIM + col] = h;
        else             Vtb[(size_t)(b * E_DIM + col) * S_DIM + (s0 + rr)] = h;
      }
    }
  }
}

// ---------------------------------------------------------------------------
// Kernel 3: flash attention, fixed-shift softmax p = exp(s - 12).
// K/Vt tiles staged in LDS via async global->LDS copies (ASYNCcnt),
// double-buffered, shared by the 4 waves of each key-half.  grid 256 x 256.
// ---------------------------------------------------------------------------
__global__ void flash_kernel(const _Float16* __restrict__ Qb,
                             const _Float16* __restrict__ Kb,
                             const _Float16* __restrict__ Vtb,
                             float* __restrict__ av) {
  __shared__ __align__(16) _Float16 Pl[8][16 * 32];        // 8 KB P tiles
  __shared__ __align__(16) _Float16 Klds[2][2][32 * 64];   // 16 KB (half,buf)
  __shared__ __align__(16) _Float16 Vlds[2][2][64 * 32];   // 16 KB (half,buf)
  __shared__ float Om[4][1024];                            // 16 KB merge
  __shared__ float Lm[4][16];

  int wid = threadIdx.x >> 5;
  int lane = threadIdx.x & 31;
  int ln = lane & 15;
  int sel = lane >> 4;
  int job = blockIdx.x * 8 + wid;   // 0..2047
  int tile = job >> 1;
  int half = job & 1;
  int wih = wid >> 1;               // wave index within its key-half group (0..3)
  int b = tile >> 8;
  int q0 = (tile & 255) * 16;
  _Float16* P = Pl[wid];

  const _Float16* qrow = Qb + (size_t)(b * S_DIM + q0 + ln) * E_DIM;
  v16h qf0 = load_frag16(qrow, 0, sel);
  v16h qf1 = load_frag16(qrow, 32, sel);

  const _Float16* Kbase = Kb + (size_t)b * S_DIM * E_DIM;
  const _Float16* Vtbase = Vtb + (size_t)b * E_DIM * S_DIM;
  uint32_t ldsK0 = lds_offset(&Klds[half][0][0]);
  uint32_t ldsV0 = lds_offset(&Vlds[half][0][0]);

  v8f o[4] = {};
  float lsum[8];
#pragma unroll
  for (int r = 0; r < 8; ++r) lsum[r] = 0.0f;
  const float SHIFT = 12.0f;

  int kbeg = half * 2048;
  const int NSTEP = 64;             // 2048 keys / 32

  // issue async fill of (buf, key0): K tile is 4KB contiguous; Vt tile is
  // 64 rows x 64B with 8KB stride. 4 waves x 32 lanes x 16B x 2 = 4KB each.
  auto issue_tiles = [&](int buf, int key0) {
    const char* gK = (const char*)(Kbase + (size_t)key0 * E_DIM);
#pragma unroll
    for (int i = 0; i < 2; ++i) {
      uint32_t off = (uint32_t)(wih * 1024 + i * 512 + lane * 16);
      async_copy16(ldsK0 + buf * 4096 + off, (uint64_t)(uintptr_t)(gK + off));
    }
#pragma unroll
    for (int i = 0; i < 2; ++i) {
      int chunk = wih * 64 + i * 32 + lane;   // 0..255
      int d = chunk >> 2;
      int part = chunk & 3;
      const char* g = (const char*)(Vtbase + (size_t)d * S_DIM + key0) + part * 16;
      async_copy16(ldsV0 + buf * 4096 + (uint32_t)(d * 64 + part * 16),
                   (uint64_t)(uintptr_t)g);
    }
  };

  issue_tiles(0, kbeg);
  wait_async0();
  __syncthreads();

  for (int st = 0; st < NSTEP; ++st) {
    int cur = st & 1;
    if (st + 1 < NSTEP) issue_tiles(cur ^ 1, kbeg + (st + 1) * 32);

    const _Float16* Kl = &Klds[half][cur][0];
    const _Float16* Vl = &Vlds[half][cur][0];

    v8f s0v = {}, s1v = {};
    {
      const _Float16* krow_lo = Kl + ln * 64;
      const _Float16* krow_hi = Kl + (16 + ln) * 64;
      v16h bl0 = load_frag16(krow_lo, 0, sel);
      v16h bl1 = load_frag16(krow_lo, 32, sel);
      v16h bh0 = load_frag16(krow_hi, 0, sel);
      v16h bh1 = load_frag16(krow_hi, 32, sel);
      s0v = wmma_f16(qf0, bl0, s0v);
      s0v = wmma_f16(qf1, bl1, s0v);
      s1v = wmma_f16(qf0, bh0, s1v);
      s1v = wmma_f16(qf1, bh1, s1v);
    }

#pragma unroll
    for (int r = 0; r < 8; ++r) {
      float p0 = __expf(s0v[r] - SHIFT);
      float p1 = __expf(s1v[r] - SHIFT);
      lsum[r] += p0 + p1;
      int rr = r + 8 * sel;
      P[rr * 32 + ln]      = (_Float16)p0;
      P[rr * 32 + 16 + ln] = (_Float16)p1;
    }
    asm volatile("s_wait_dscnt 0x0" ::: "memory");
    v16h pf;
    {
      const v8h lo = *(const v8h*)(P + ln * 32 + sel * 8);
      const v8h hi = *(const v8h*)(P + ln * 32 + 16 + sel * 8);
#pragma unroll
      for (int i = 0; i < 8; ++i) { pf[i] = lo[i]; pf[8 + i] = hi[i]; }
    }
#pragma unroll
    for (int nt = 0; nt < 4; ++nt) {
      const _Float16* vrow = Vl + (nt * 16 + ln) * 32;
      v16h bv = load_frag16(vrow, 0, sel);
      o[nt] = wmma_f16(pf, bv, o[nt]);
    }

    wait_async0();     // my next-buffer fills done
    __syncthreads();   // everyone's fills done / everyone finished reading cur
  }

  // one-time cross-lane row sums (16-lane butterfly stays in the half)
#pragma unroll
  for (int r = 0; r < 8; ++r) {
#pragma unroll
    for (int off = 1; off < 16; off <<= 1)
      lsum[r] += __shfl_xor(lsum[r], off, 32);
  }

  // merge the two key-halves (wave pairs wid, wid^1)
  int pr = wid >> 1;
  if (half == 1) {
    float* od = Om[pr];
#pragma unroll
    for (int nt = 0; nt < 4; ++nt) {
      int col = nt * 16 + ln;
#pragma unroll
      for (int r = 0; r < 8; ++r) od[(r + 8 * sel) * 64 + col] = o[nt][r];
    }
    if (ln == 0) {
#pragma unroll
      for (int r = 0; r < 8; ++r) Lm[pr][r + 8 * sel] = lsum[r];
    }
  }
  __syncthreads();
  if (half == 0) {
    const float* od = Om[pr];
    float* avbase = av + (size_t)(b * S_DIM + q0) * E_DIM;
#pragma unroll
    for (int nt = 0; nt < 4; ++nt) {
      int col = nt * 16 + ln;
#pragma unroll
      for (int r = 0; r < 8; ++r) {
        int rr = r + 8 * sel;
        float num = o[nt][r] + od[rr * 64 + col];
        float den = lsum[r] + Lm[pr][rr];
        avbase[(size_t)rr * 64 + col] = num / den;
      }
    }
  }
}

// ---------------------------------------------------------------------------
// Kernel 4: final = sum(av[b]*coeffs) + offset
// ---------------------------------------------------------------------------
__global__ void final_kernel(const float* __restrict__ av,
                             const float* __restrict__ coeffs,
                             const float* __restrict__ offset,
                             float* __restrict__ out) {
  __shared__ float red[256];
  int b = blockIdx.x;
  int tid = threadIdx.x;
  const float* a = av + (size_t)b * S_DIM * E_DIM;
  float s = 0.0f;
  for (int i = tid; i < S_DIM * E_DIM; i += 256) s += a[i] * coeffs[i];
  red[tid] = s;
  __syncthreads();
  for (int st = 128; st > 0; st >>= 1) {
    if (tid < st) red[tid] += red[tid + st];
    __syncthreads();
  }
  if (tid == 0) out[b] = red[0] + offset[0];
}

// ---------------------------------------------------------------------------
extern "C" void kernel_launch(void* const* d_in, const int* in_sizes, int n_in,
                              void* d_out, int out_size, void* d_ws, size_t ws_size,
                              hipStream_t stream) {
  const float* x      = (const float*)d_in[0];
  const float* qw     = (const float*)d_in[1];
  const float* kw     = (const float*)d_in[2];
  const float* vw     = (const float*)d_in[3];
  const float* rm     = (const float*)d_in[4];
  const float* coeffs = (const float*)d_in[5];
  const float* offset = (const float*)d_in[6];
  float* out = (float*)d_out;

  char* ws = (char*)d_ws;
  float*    av  = (float*)ws;                                 // 4 MB
  _Float16* Qb  = (_Float16*)(ws + ((size_t)4 << 20));        // 2 MB
  _Float16* Kb  = (_Float16*)(ws + ((size_t)6 << 20));        // 2 MB
  _Float16* Vtb = (_Float16*)(ws + ((size_t)8 << 20));        // 2 MB
  _Float16* rmT = (_Float16*)(ws + ((size_t)10 << 20));       // 512 KB
  _Float16* wT  = (_Float16*)(ws + ((size_t)10 << 20) + (1 << 19));

  prep_kernel<<<1024, 256, 0, stream>>>(rm, qw, kw, vw, rmT, wT);
  proj_kernel<<<512, 256, 0, stream>>>(x, rmT, av);
  for (int l = 0; l < 3; ++l) {
    qkv_kernel<<<128, 256, 0, stream>>>(av, wT, l, Qb, Kb, Vtb);
    flash_kernel<<<256, 256, 0, stream>>>(Qb, Kb, Vtb, av);
  }
  final_kernel<<<4, 256, 0, stream>>>(av, coeffs, offset, out);
}
